// Block_78984448573482
// MI455X (gfx1250) — compile-verified, matrix operand order checked by reading, not run on hardware
//
#include <hip/hip_runtime.h>

#define B_  4
#define T_  1024
#define H_  1024
#define NH_ 16
#define HS_ 64
#define FF_ 4096

typedef __attribute__((ext_vector_type(16))) __bf16          v16bf;
typedef __attribute__((ext_vector_type(8)))  float           v8f;
typedef __attribute__((ext_vector_type(8)))  unsigned short  ushort8v;
typedef __attribute__((ext_vector_type(4)))  unsigned int    v4u;
typedef __attribute__((ext_vector_type(8)))  int             v8i;
typedef __attribute__((ext_vector_type(4)))  int             v4i;

__device__ __forceinline__ unsigned short f2bf(float f) {
    unsigned int u = __float_as_uint(f);
    return (unsigned short)((u + 0x7fffu + ((u >> 16) & 1u)) >> 16);  // RNE
}

// Issue one TDM 2D tile load (group0 built here, shared group1 passed in).
// D# group0: [1:0]=count=1, [63:32]=lds_addr, [120:64]=global_addr, [127:126]=type=2
__device__ __forceinline__ void tdm_issue(unsigned lds, const unsigned short* gaddr, v8i g1)
{
    unsigned long long ga = (unsigned long long)(const void*)gaddr;
    v4u g0;
    g0[0] = 1u;                                                // count = 1 valid descriptor
    g0[1] = lds;                                               // LDS byte address
    g0[2] = (unsigned)ga;                                      // global_addr[31:0]
    g0[3] = ((unsigned)(ga >> 32) & 0x01ffffffu) | 0x80000000u; // global_addr[56:32] | type=2
    v4i z4 = (v4i)0;
#if __clang_major__ >= 23
    v8i z8 = (v8i)0;
    __builtin_amdgcn_tensor_load_to_lds(g0, g1, z4, z4, z8, 0);
#else
    __builtin_amdgcn_tensor_load_to_lds(g0, g1, z4, z4, 0);
#endif
}

// ---------------------------------------------------------------------------
// Batched GEMM: C[M,N] = scale * (A[M,K] x Bt[N,K]^T) + bias + resid
// A, Bt bf16 (ushort); C fp32. 64x64 block tile, 128 threads (4 waves),
// each wave a 32x32 sub-tile -> 4x v_wmma_f32_16x16x32_bf16 per 32-deep
// K step. Tiles staged global->LDS by the Tensor Data Mover (wave 0 posts
// descriptors; TENSORcnt double-buffer pipeline overlaps DMA with WMMA).
// TDM pad feature (16 DWORDs interval, +4 DWORDs) creates the 40-short LDS
// row pitch (16B-aligned fragment reads, bank-conflict-free).
// ---------------------------------------------------------------------------
__global__ void __launch_bounds__(128)
gemm_wmma_bf16(const unsigned short* __restrict__ A,
               const unsigned short* __restrict__ Bt,
               float* __restrict__ C,
               const float* __restrict__ bias,
               const float* __restrict__ resid,
               int M, int N, int K,
               long long sA, long long sB, long long sC,
               float scale)
{
    __shared__ unsigned short As[2][64 * 40];   // double-buffered, 40-short pitch
    __shared__ unsigned short Bs[2][64 * 40];

    const unsigned short* Ab = A  + (size_t)blockIdx.z * sA;
    const unsigned short* Bb = Bt + (size_t)blockIdx.z * sB;
    float*                Cb = C  + (size_t)blockIdx.z * sC;

    const int rowBase = blockIdx.y * 64;
    const int colBase = blockIdx.x * 64;

    const int tid   = threadIdx.x;
    const int lane  = tid & 31;
    const int wave  = tid >> 5;
    const int waveM = wave >> 1;          // 0..1
    const int waveN = wave & 1;           // 0..1
    const int halfw = lane >> 4;          // 0..1 (selects K half / M+8)
    const int lrow  = lane & 15;
    const int kb    = halfw * 8;

    // Shared D# group1: data_size=2B, pad_enable, pad_interval=3 (16 DW),
    // pad_amount=3 (+4 DW); tensor 32 x 64, tile 32 x 64, dim0 stride = K.
    v8i g1;
    g1[0] = 0x00010000 | (1 << 20) | (3 << 22) | (3 << 25);
    g1[1] = 32 << 16;          // tensor_dim0[15:0] in bits 63:48
    g1[2] = 64 << 16;          // tensor_dim0 hi = 0 ; tensor_dim1[15:0]
    g1[3] = 32 << 16;          // tensor_dim1 hi = 0 ; tile_dim0 = 32
    g1[4] = 64;                // tile_dim1 = 64, tile_dim2 = 0
    g1[5] = K;                 // tensor_dim0_stride[31:0]
    g1[6] = 0;                 // stride hi / dim1_stride
    g1[7] = 0;

    const int nk = K / 32;

    if (wave == 0)  {          // preload tile 0 via TDM
        tdm_issue((unsigned)(size_t)&As[0][0], Ab + (size_t)rowBase * K, g1);
        tdm_issue((unsigned)(size_t)&Bs[0][0], Bb + (size_t)colBase * K, g1);
    }

    v8f c00 = {}, c01 = {}, c10 = {}, c11 = {};

    for (int kt = 0; kt < nk; ++kt) {
        const int buf = kt & 1;
        if (wave == 0) {
            if (kt + 1 < nk) {   // post next tile, then wait for current (in-order TENSORcnt)
                tdm_issue((unsigned)(size_t)&As[buf ^ 1][0],
                          Ab + (size_t)rowBase * K + (kt + 1) * 32, g1);
                tdm_issue((unsigned)(size_t)&Bs[buf ^ 1][0],
                          Bb + (size_t)colBase * K + (kt + 1) * 32, g1);
                __builtin_amdgcn_s_wait_tensorcnt(2);
            } else {
                __builtin_amdgcn_s_wait_tensorcnt(0);
            }
        }
        __syncthreads();

        union Frag { v16bf v; ushort8v h[2]; } a0, a1, b0, b1;
        const unsigned short* pa0 = &As[buf][(waveM * 32 +  0 + lrow) * 40 + kb];
        const unsigned short* pa1 = &As[buf][(waveM * 32 + 16 + lrow) * 40 + kb];
        const unsigned short* pb0 = &Bs[buf][(waveN * 32 +  0 + lrow) * 40 + kb];
        const unsigned short* pb1 = &Bs[buf][(waveN * 32 + 16 + lrow) * 40 + kb];
        a0.h[0] = *(const ushort8v*)pa0;  a0.h[1] = *(const ushort8v*)(pa0 + 16);
        a1.h[0] = *(const ushort8v*)pa1;  a1.h[1] = *(const ushort8v*)(pa1 + 16);
        b0.h[0] = *(const ushort8v*)pb0;  b0.h[1] = *(const ushort8v*)(pb0 + 16);
        b1.h[0] = *(const ushort8v*)pb1;  b1.h[1] = *(const ushort8v*)(pb1 + 16);

        c00 = __builtin_amdgcn_wmma_f32_16x16x32_bf16(false, a0.v, false, b0.v, (short)0, c00, false, false);
        c01 = __builtin_amdgcn_wmma_f32_16x16x32_bf16(false, a0.v, false, b1.v, (short)0, c01, false, false);
        c10 = __builtin_amdgcn_wmma_f32_16x16x32_bf16(false, a1.v, false, b0.v, (short)0, c10, false, false);
        c11 = __builtin_amdgcn_wmma_f32_16x16x32_bf16(false, a1.v, false, b1.v, (short)0, c11, false, false);
        __syncthreads();
    }

    #pragma unroll
    for (int i = 0; i < 2; i++) {
        #pragma unroll
        for (int j = 0; j < 2; j++) {
            v8f cc = (i == 0) ? (j == 0 ? c00 : c01) : (j == 0 ? c10 : c11);
            const int gm0 = rowBase + waveM * 32 + i * 16 + halfw * 8;
            const int gn  = colBase + waveN * 32 + j * 16 + lrow;
            const float badd = bias ? bias[gn] : 0.0f;
            #pragma unroll
            for (int r = 0; r < 8; r++) {
                size_t idx = (size_t)(gm0 + r) * N + gn;
                float v = cc[r] * scale + badd;
                if (resid) v += resid[idx];
                Cb[idx] = v;
            }
        }
    }
}

// ---------------------------------------------------------------------------
// LayerNorm over H=1024, fp32 in -> bf16 out (GEMM A operand)
// ---------------------------------------------------------------------------
__global__ void __launch_bounds__(256)
layernorm_bf16(const float* __restrict__ x, const float* __restrict__ w,
               const float* __restrict__ b, unsigned short* __restrict__ out)
{
    __shared__ float red[256];
    const int row = blockIdx.x;
    const float* xr = x + (size_t)row * H_;
    float v[4], s = 0.f;
    #pragma unroll
    for (int i = 0; i < 4; i++) { v[i] = xr[threadIdx.x + i * 256]; s += v[i]; }
    red[threadIdx.x] = s; __syncthreads();
    for (int o = 128; o > 0; o >>= 1) {
        if (threadIdx.x < o) red[threadIdx.x] += red[threadIdx.x + o];
        __syncthreads();
    }
    const float mu = red[0] * (1.0f / H_);
    __syncthreads();
    s = 0.f;
    #pragma unroll
    for (int i = 0; i < 4; i++) { float d = v[i] - mu; s += d * d; }
    red[threadIdx.x] = s; __syncthreads();
    for (int o = 128; o > 0; o >>= 1) {
        if (threadIdx.x < o) red[threadIdx.x] += red[threadIdx.x + o];
        __syncthreads();
    }
    const float rstd = rsqrtf(red[0] * (1.0f / H_) + 1e-5f);
    unsigned short* orow = out + (size_t)row * H_;
    #pragma unroll
    for (int i = 0; i < 4; i++) {
        int c = threadIdx.x + i * 256;
        orow[c] = f2bf((v[i] - mu) * rstd * w[c] + b[c]);
    }
}

// fp32 W[K,N] -> bf16 Wt[N,K]  (GEMM B operand layout, coalesced K reads)
__global__ void convert_transpose(const float* __restrict__ W,
                                  unsigned short* __restrict__ Wt, int K, int N)
{
    int idx = blockIdx.x * blockDim.x + threadIdx.x;  // over N*K, out-linear
    int k = idx % K, n = idx / K;
    Wt[idx] = f2bf(W[(size_t)k * N + n]);
}

// [B,T,NH,HS] fp32 -> RoPE -> [B,NH,T,HS] bf16 (interleaved-pair, base 10000)
__global__ void rope_to_heads(const float* __restrict__ in,
                              unsigned short* __restrict__ out)
{
    int idx = blockIdx.x * blockDim.x + threadIdx.x;  // B*T*NH*32 pairs
    int i  = idx & 31;
    int nh = (idx >> 5) & 15;
    int t  = (idx >> 9) & 1023;
    int b  = idx >> 19;
    const float* p = in + ((((size_t)b * T_ + t) * NH_ + nh) * HS_ + 2 * i);
    float x0 = p[0], x1 = p[1];
    float ang = (float)t * __powf(10000.f, -(float)i / 32.f);
    float sn, cs; __sincosf(ang, &sn, &cs);
    unsigned short* q = out + ((((size_t)b * NH_ + nh) * T_ + t) * HS_ + 2 * i);
    q[0] = f2bf(x0 * cs - x1 * sn);
    q[1] = f2bf(x0 * sn + x1 * cs);
}

// [B,T,NH,HS] fp32 -> [B,NH,HS,T] bf16   (V^T so attn@V GEMM reads [N=HS][K=T])
__global__ void v_transpose(const float* __restrict__ in,
                            unsigned short* __restrict__ out)
{
    int idx = blockIdx.x * blockDim.x + threadIdx.x;  // out-linear, B*NH*HS*T
    int t  = idx & 1023;
    int hs = (idx >> 10) & 63;
    int nh = (idx >> 16) & 15;
    int b  = idx >> 20;
    out[idx] = f2bf(in[(((size_t)b * T_ + t) * NH_ + nh) * HS_ + hs]);
}

// softmax over full row, THEN multiplicative causal mask (no renorm) -> bf16
__global__ void __launch_bounds__(256)
softmax_postmask(const float* __restrict__ sc, unsigned short* __restrict__ attn)
{
    __shared__ float red[256];
    const int qi = blockIdx.x, bh = blockIdx.y;
    const float* row = sc + ((size_t)bh * T_ + qi) * T_;
    float v[4], m = -1e30f;
    #pragma unroll
    for (int i = 0; i < 4; i++) { v[i] = row[threadIdx.x + i * 256]; m = fmaxf(m, v[i]); }
    red[threadIdx.x] = m; __syncthreads();
    for (int o = 128; o > 0; o >>= 1) {
        if (threadIdx.x < o) red[threadIdx.x] = fmaxf(red[threadIdx.x], red[threadIdx.x + o]);
        __syncthreads();
    }
    m = red[0]; __syncthreads();
    float e[4], s = 0.f;
    #pragma unroll
    for (int i = 0; i < 4; i++) { e[i] = __expf(v[i] - m); s += e[i]; }
    red[threadIdx.x] = s; __syncthreads();
    for (int o = 128; o > 0; o >>= 1) {
        if (threadIdx.x < o) red[threadIdx.x] += red[threadIdx.x + o];
        __syncthreads();
    }
    const float inv = 1.0f / red[0];
    unsigned short* orow = attn + ((size_t)bh * T_ + qi) * T_;
    #pragma unroll
    for (int i = 0; i < 4; i++) {
        int c = threadIdx.x + i * 256;
        orow[c] = (c <= qi) ? f2bf(e[i] * inv) : (unsigned short)0;
    }
}

// [B,NH,T,HS] fp32 -> [B,T,H] bf16 (merge heads for o-projection)
__global__ void merge_heads(const float* __restrict__ in,
                            unsigned short* __restrict__ out)
{
    int idx = blockIdx.x * blockDim.x + threadIdx.x;  // out-linear, B*T*H
    int hc = idx & 1023;
    int t  = (idx >> 10) & 1023;
    int b  = idx >> 20;
    int nh = hc >> 6, hs = hc & 63;
    out[idx] = f2bf(in[(((size_t)b * NH_ + nh) * T_ + t) * HS_ + hs]);
}

// exact GELU, fp32 -> bf16
__global__ void gelu_bf16(const float* __restrict__ in,
                          unsigned short* __restrict__ out, int n)
{
    int idx = blockIdx.x * blockDim.x + threadIdx.x;
    if (idx < n) {
        float x = in[idx];
        out[idx] = f2bf(0.5f * x * (1.0f + erff(x * 0.70710678118f)));
    }
}

// ---------------------------------------------------------------------------
extern "C" void kernel_launch(void* const* d_in, const int* in_sizes, int n_in,
                              void* d_out, int out_size, void* d_ws, size_t ws_size,
                              hipStream_t stream)
{
    (void)in_sizes; (void)n_in; (void)out_size; (void)ws_size;
    const float* x    = (const float*)d_in[0];
    const float* Wq   = (const float*)d_in[1];
    const float* bq   = (const float*)d_in[2];
    const float* Wk   = (const float*)d_in[3];
    const float* bk   = (const float*)d_in[4];
    const float* Wv   = (const float*)d_in[5];
    const float* bv   = (const float*)d_in[6];
    const float* Wo   = (const float*)d_in[7];
    const float* bo   = (const float*)d_in[8];
    const float* ln1w = (const float*)d_in[9];
    const float* ln1b = (const float*)d_in[10];
    const float* ln2w = (const float*)d_in[11];
    const float* ln2b = (const float*)d_in[12];
    const float* Wfc  = (const float*)d_in[13];
    const float* bfc  = (const float*)d_in[14];
    const float* Wpr  = (const float*)d_in[15];
    const float* bpr  = (const float*)d_in[16];
    float* out = (float*)d_out;

    char* ws = (char*)d_ws;
    size_t off = 0;
    auto alloc = [&](size_t bytes) -> void* {
        off = (off + 255) & ~(size_t)255;
        void* p = ws + off;
        off += bytes;
        return p;
    };

    const int MT = B_ * T_;                                    // 4096 rows
    unsigned short* wqT  = (unsigned short*)alloc((size_t)H_ * H_ * 2);
    unsigned short* wkT  = (unsigned short*)alloc((size_t)H_ * H_ * 2);
    unsigned short* wvT  = (unsigned short*)alloc((size_t)H_ * H_ * 2);
    unsigned short* woT  = (unsigned short*)alloc((size_t)H_ * H_ * 2);
    unsigned short* wfcT = (unsigned short*)alloc((size_t)FF_ * H_ * 2);
    unsigned short* wprT = (unsigned short*)alloc((size_t)H_ * FF_ * 2);
    unsigned short* h1   = (unsigned short*)alloc((size_t)MT * H_ * 2);
    float* qf  = (float*)alloc((size_t)MT * H_ * 4);
    float* kf  = (float*)alloc((size_t)MT * H_ * 4);
    float* vf  = (float*)alloc((size_t)MT * H_ * 4);
    unsigned short* qb  = (unsigned short*)alloc((size_t)MT * H_ * 2);
    unsigned short* kbb = (unsigned short*)alloc((size_t)MT * H_ * 2);
    unsigned short* vtb = (unsigned short*)alloc((size_t)MT * H_ * 2);
    float* sc = (float*)alloc((size_t)B_ * NH_ * T_ * T_ * 4);            // 256 MB
    unsigned short* attn = (unsigned short*)alloc((size_t)B_ * NH_ * T_ * T_ * 2); // 128 MB
    float* aof = (float*)alloc((size_t)MT * H_ * 4);
    unsigned short* aob = (unsigned short*)alloc((size_t)MT * H_ * 2);
    float* xres = (float*)alloc((size_t)MT * H_ * 4);
    unsigned short* h2 = (unsigned short*)alloc((size_t)MT * H_ * 2);
    float* fcf = (float*)alloc((size_t)MT * FF_ * 4);                     // 64 MB
    unsigned short* mb = (unsigned short*)alloc((size_t)MT * FF_ * 2);

    // --- weight convert + transpose (bf16, [N,K]) ---
    convert_transpose<<<(H_ * H_) / 256, 256, 0, stream>>>(Wq,  wqT,  H_,  H_);
    convert_transpose<<<(H_ * H_) / 256, 256, 0, stream>>>(Wk,  wkT,  H_,  H_);
    convert_transpose<<<(H_ * H_) / 256, 256, 0, stream>>>(Wv,  wvT,  H_,  H_);
    convert_transpose<<<(H_ * H_) / 256, 256, 0, stream>>>(Wo,  woT,  H_,  H_);
    convert_transpose<<<(H_ * FF_) / 256, 256, 0, stream>>>(Wfc, wfcT, H_,  FF_);
    convert_transpose<<<(FF_ * H_) / 256, 256, 0, stream>>>(Wpr, wprT, FF_, H_);

    // --- attention sub-block ---
    layernorm_bf16<<<MT, 256, 0, stream>>>(x, ln1w, ln1b, h1);

    dim3 gq(H_ / 64, MT / 64, 1);
    gemm_wmma_bf16<<<gq, 128, 0, stream>>>(h1, wqT, qf, bq, nullptr, MT, H_, H_, 0, 0, 0, 1.0f);
    gemm_wmma_bf16<<<gq, 128, 0, stream>>>(h1, wkT, kf, bk, nullptr, MT, H_, H_, 0, 0, 0, 1.0f);
    gemm_wmma_bf16<<<gq, 128, 0, stream>>>(h1, wvT, vf, bv, nullptr, MT, H_, H_, 0, 0, 0, 1.0f);

    rope_to_heads<<<(B_ * T_ * NH_ * 32) / 256, 256, 0, stream>>>(qf, qb);
    rope_to_heads<<<(B_ * T_ * NH_ * 32) / 256, 256, 0, stream>>>(kf, kbb);
    v_transpose<<<(B_ * NH_ * HS_ * T_) / 256, 256, 0, stream>>>(vf, vtb);

    // scores = (q @ k^T) / sqrt(64), per (b,h) batch
    dim3 gs(T_ / 64, T_ / 64, B_ * NH_);
    gemm_wmma_bf16<<<gs, 128, 0, stream>>>(qb, kbb, sc, nullptr, nullptr,
        T_, T_, HS_, (long long)T_ * HS_, (long long)T_ * HS_, (long long)T_ * T_, 0.125f);

    dim3 gsm(T_, B_ * NH_);
    softmax_postmask<<<gsm, 256, 0, stream>>>(sc, attn);

    // out = attn @ v   (Bt = V^T [HS,T])
    dim3 go(HS_ / 64, T_ / 64, B_ * NH_);
    gemm_wmma_bf16<<<go, 128, 0, stream>>>(attn, vtb, aof, nullptr, nullptr,
        T_, HS_, T_, (long long)T_ * T_, (long long)HS_ * T_, (long long)T_ * HS_, 1.0f);

    merge_heads<<<(B_ * T_ * H_) / 256, 256, 0, stream>>>(aof, aob);

    // x_res = x + out @ Wo + bo
    gemm_wmma_bf16<<<gq, 128, 0, stream>>>(aob, woT, xres, bo, x, MT, H_, H_, 0, 0, 0, 1.0f);

    // --- MLP sub-block ---
    layernorm_bf16<<<MT, 256, 0, stream>>>(xres, ln2w, ln2b, h2);

    dim3 gfc(FF_ / 64, MT / 64, 1);
    gemm_wmma_bf16<<<gfc, 128, 0, stream>>>(h2, wfcT, fcf, bfc, nullptr, MT, FF_, H_, 0, 0, 0, 1.0f);

    gelu_bf16<<<(MT * FF_) / 256, 256, 0, stream>>>(fcf, mb, MT * FF_);

    // out = x_res + m @ Wpr + bpr   (written straight to d_out)
    gemm_wmma_bf16<<<gq, 128, 0, stream>>>(mb, wprT, out, bpr, xres, MT, H_, FF_, 0, 0, 0, 1.0f);
}
